// MixtralAttention_2886218023294
// MI455X (gfx1250) — compile-verified, matrix operand order checked by reading, not run on hardware
//
#include <hip/hip_runtime.h>

// ---------------------------------------------------------------------------
// Mixtral attention prefill (T=2048) for gfx1250 / MI455X.
// Pipeline:
//   0) one-shot fp32 -> bf16 conversion of hidden / w_qkv / w_o
//   1) QKV GEMM  (bf16 in, WMMA, async-to-LDS stage-in)
//   2) RoPE (fp32, in place on qkv)
//   3) causal flash attention (bf16 WMMA, writes bf16)
//   4) O-proj GEMM (bf16 in, WMMA, fp32 out -> d_out)
// Workspace: hidden_bf + wqkv_bf + wo_bf + qkv_f32 + attn_bf  (~168 MB)
// ---------------------------------------------------------------------------

#define HIDDEN    4096
#define QKV_ROW   6144   // Q_SIZE + 2*KV_SIZE
#define QS        4096   // Q_SIZE
#define KVS       1024   // KV_SIZE
#define HEAD_DIM  128
#define NUM_HEADS 32
#define NUM_KV    8

typedef __attribute__((ext_vector_type(16))) __bf16        v16bf;
typedef __attribute__((ext_vector_type(8)))  float         v8f;
typedef __attribute__((ext_vector_type(8)))  unsigned int  v8u;

// 8 args: (neg_a, A, neg_b, B, c_mod, C, reuse_a, reuse_b)
#define WMMA_BF16(A, B, C) \
  __builtin_amdgcn_wmma_f32_16x16x32_bf16(false, (A), false, (B), (short)0, (C), false, false)

#if defined(__AMDGCN__) && __has_builtin(__builtin_amdgcn_global_load_async_to_lds_b128)
#define ASYNC_CP 1
#endif

__device__ __forceinline__ unsigned int bf16_rne(float f) {
  unsigned int u = __float_as_uint(f);
  return (u + 0x7fffu + ((u >> 16) & 1u)) >> 16;
}
__device__ __forceinline__ unsigned int pack2_bf16(float lo, float hi) {
  return bf16_rne(lo) | (bf16_rne(hi) << 16);
}
__device__ __forceinline__ v16bf frag8(const unsigned int* r) {
  v8u t;
#pragma unroll
  for (int i = 0; i < 8; ++i) t[i] = r[i];
  return __builtin_bit_cast(v16bf, t);
}

// 16-byte global -> LDS copy (async DMA path when available).
// Builtin parameter type (from hipcc diagnostic): int __vector(4) * with
// addrspace(1) global source / addrspace(3) LDS destination.
#if defined(ASYNC_CP)
typedef int b128_t __attribute__((vector_size(16)));
typedef __attribute__((address_space(1))) b128_t* g_b128_p;
typedef __attribute__((address_space(3))) b128_t* l_b128_p;
#endif

__device__ __forceinline__ void cp16_g2l(const void* g, void* l) {
#if defined(ASYNC_CP)
  __builtin_amdgcn_global_load_async_to_lds_b128(
      (g_b128_p)(size_t)g,
      (l_b128_p)(unsigned)(size_t)l,
      0, 0);
#else
  *(uint4*)l = *(const uint4*)g;
#endif
}
__device__ __forceinline__ void cp16_wait() {
#if defined(ASYNC_CP)
#if __has_builtin(__builtin_amdgcn_s_wait_asynccnt)
  __builtin_amdgcn_s_wait_asynccnt(0);
#else
  asm volatile("s_wait_asynccnt 0" ::: "memory");
#endif
#endif
}

// ---------------------------------------------------------------------------
// One-shot fp32 -> packed bf16 conversion (2 elements per thread)
// ---------------------------------------------------------------------------
__global__ void cvt_bf16_kernel(const float* __restrict__ src,
                                unsigned int* __restrict__ dst, int npairs) {
  int i = blockIdx.x * blockDim.x + threadIdx.x;
  if (i >= npairs) return;
  float2 v = ((const float2*)src)[i];
  dst[i] = pack2_bf16(v.x, v.y);
}

// ---------------------------------------------------------------------------
// C[M,N] (fp32) = A[M,K] (bf16) * B[N,K]^T (bf16)
// Block: 256 threads = 8 waves, tile 128x128, k-step 32.
// Wave grid 2(M) x 4(N): each wave owns 64x32 = 4x2 WMMA tiles.
// Stage-in: GLOBAL_LOAD_ASYNC_TO_LDS_B128 (ASYNCcnt) when available.
// LDS rows: 64B of data + 16B pad = 80B (16B-aligned for B128,
// 20-uint stride => conflict-free fragment reads).
// ---------------------------------------------------------------------------
__global__ __launch_bounds__(256) void gemm_bf16_nt(
    const unsigned short* __restrict__ A, const unsigned short* __restrict__ B,
    float* __restrict__ C, int M, int N, int K) {
  __shared__ __align__(16) unsigned int lA[128 * 20];
  __shared__ __align__(16) unsigned int lB[128 * 20];

  const int tid  = threadIdx.x;
  const int lane = tid & 31;
  const int wave = tid >> 5;
  const int wm   = wave >> 2;   // 0..1
  const int wn   = wave & 3;    // 0..3
  const int half = lane >> 4;
  const int l15  = lane & 15;
  const int m0   = blockIdx.y * 128;
  const int n0   = blockIdx.x * 128;

  v8f acc[4][2] = {};

  for (int k0 = 0; k0 < K; k0 += 32) {
    __syncthreads();
    // Stage A/B tiles: 128 rows x 64B each = 512 B128 transfers per tile,
    // 2 per thread per tile.
#pragma unroll
    for (int i = 0; i < 2; ++i) {
      int idx = tid + i * 256;            // 0..511
      int r = idx >> 2, seg = idx & 3;    // row, 16B segment
      cp16_g2l((const char*)A + ((size_t)(m0 + r) * K + k0) * 2 + seg * 16,
               (char*)lA + r * 80 + seg * 16);
    }
#pragma unroll
    for (int i = 0; i < 2; ++i) {
      int idx = tid + i * 256;
      int r = idx >> 2, seg = idx & 3;
      cp16_g2l((const char*)B + ((size_t)(n0 + r) * K + k0) * 2 + seg * 16,
               (char*)lB + r * 80 + seg * 16);
    }
    cp16_wait();
    __syncthreads();

    // Fragment loads follow CDNA5 16-bit A/B layout:
    // VGPR p holds k-pair kp = (p<4?0:8) + half*4 + (p&3)
    unsigned int ar[4][8], br[2][8];
#pragma unroll
    for (int mt = 0; mt < 4; ++mt) {
      int m = wm * 64 + mt * 16 + l15;
#pragma unroll
      for (int p = 0; p < 8; ++p) {
        int kp = (p < 4 ? 0 : 8) + half * 4 + (p & 3);
        ar[mt][p] = lA[m * 20 + kp];
      }
    }
#pragma unroll
    for (int nt = 0; nt < 2; ++nt) {
      int n = wn * 32 + nt * 16 + l15;
#pragma unroll
      for (int p = 0; p < 8; ++p) {
        int kp = (p < 4 ? 0 : 8) + half * 4 + (p & 3);
        br[nt][p] = lB[n * 20 + kp];
      }
    }
#pragma unroll
    for (int mt = 0; mt < 4; ++mt) {
      v16bf a = frag8(ar[mt]);
#pragma unroll
      for (int nt = 0; nt < 2; ++nt) {
        acc[mt][nt] = WMMA_BF16(a, frag8(br[nt]), acc[mt][nt]);
      }
    }
  }

  // C/D layout: vgpr r -> row (r + 8*half), lane&15 -> col
#pragma unroll
  for (int mt = 0; mt < 4; ++mt)
#pragma unroll
    for (int nt = 0; nt < 2; ++nt)
#pragma unroll
      for (int r = 0; r < 8; ++r) {
        int m = m0 + wm * 64 + mt * 16 + r + 8 * half;
        int n = n0 + wn * 32 + nt * 16 + l15;
        C[(size_t)m * N + n] = acc[mt][nt][r];
      }
}

// ---------------------------------------------------------------------------
// RoPE on q (heads 0..31) and k (heads 32..39) portions of qkv, in place.
// ---------------------------------------------------------------------------
__global__ void rope_kernel(float* __restrict__ qkv,
                            const int* __restrict__ positions, int T) {
  int idx = blockIdx.x * blockDim.x + threadIdx.x;
  int total = T * (NUM_HEADS + NUM_KV) * (HEAD_DIM / 2);
  if (idx >= total) return;
  int i = idx & 63;
  int h = (idx >> 6) % (NUM_HEADS + NUM_KV);
  int t = idx / (64 * (NUM_HEADS + NUM_KV));
  float inv = __powf(1.0e6f, -(float)i / 64.0f);
  float ang = (float)positions[t] * inv;
  float s, c;
  __sincosf(ang, &s, &c);
  size_t base = (size_t)t * QKV_ROW +
                (h < NUM_HEADS ? (size_t)h * HEAD_DIM
                               : (size_t)QS + (size_t)(h - NUM_HEADS) * HEAD_DIM);
  float x1 = qkv[base + i];
  float x2 = qkv[base + 64 + i];
  qkv[base + i]      = x1 * c - x2 * s;
  qkv[base + 64 + i] = x2 * c + x1 * s;
}

// ---------------------------------------------------------------------------
// Flash attention: block = (head h, 64-row q block), 128 threads = 4 waves.
// Each wave owns 16 q rows; kv in 32-wide blocks, online softmax.
// QK^T and PV via bf16 WMMA. Output written as bf16 for the O-proj GEMM.
// ---------------------------------------------------------------------------
__global__ __launch_bounds__(128) void attn_kernel(
    const float* __restrict__ qkv, unsigned short* __restrict__ attn_out, int T) {
  __shared__ unsigned int  lQ[64][65];    // 64 q rows x 64 bf16-pairs (d) + pad
  __shared__ unsigned int  lK[32][65];    // 32 kv rows x 64 bf16-pairs (d)
  __shared__ unsigned int  lVt[128][17];  // V transposed: d x 16 kv-pairs
  __shared__ unsigned int  lP[4][16][17]; // per-wave probs: 16 rows x 16 kv-pairs

  const int h    = blockIdx.x;           // q head
  const int qb   = blockIdx.y;           // q block of 64
  const int g    = h >> 2;               // kv head (GROUP=4)
  const int tid  = threadIdx.x;
  const int lane = tid & 31;
  const int wave = tid >> 5;
  const int half = lane >> 4;
  const int l15  = lane & 15;
  const int q0   = qb * 64;
  const int qw   = q0 + wave * 16;

  const float* Qb = qkv + (size_t)h * HEAD_DIM;
  const float* Kb = qkv + QS + (size_t)g * HEAD_DIM;
  const float* Vb = qkv + QS + KVS + (size_t)g * HEAD_DIM;

  // Stage Q tile: 64 rows x 64 pairs = 4096 pairs / 128 thr = 32 each
#pragma unroll
  for (int i = 0; i < 32; ++i) {
    int idx = tid + i * 128;
    int r = idx >> 6, kp = idx & 63;
    const float2 v = *(const float2*)(Qb + (size_t)(q0 + r) * QKV_ROW + kp * 2);
    lQ[r][kp] = pack2_bf16(v.x, v.y);
  }
  __syncthreads();

  v8f o[8] = {};
  float mrow[8], lrow[8];
#pragma unroll
  for (int r = 0; r < 8; ++r) { mrow[r] = -3.0e38f; lrow[r] = 0.0f; }
  const float scale = 0.08838834764831845f;  // 1/sqrt(128)

  const int kv_end = q0 + 64;  // causal

  for (int j0 = 0; j0 < kv_end; j0 += 32) {
    __syncthreads();
    // Stage K block: 32 rows x 64 pairs
#pragma unroll
    for (int i = 0; i < 16; ++i) {
      int idx = tid + i * 128;
      int r = idx >> 6, kp = idx & 63;
      const float2 v = *(const float2*)(Kb + (size_t)(j0 + r) * QKV_ROW + kp * 2);
      lK[r][kp] = pack2_bf16(v.x, v.y);
    }
    // Stage V transposed (d-major) so PV B-fragments read contiguous k-pairs
#pragma unroll
    for (int i = 0; i < 16; ++i) {
      int idx = tid + i * 128;
      int r = idx >> 6, kp = idx & 63;
      const float2 v = *(const float2*)(Vb + (size_t)(j0 + r) * QKV_ROW + kp * 2);
      ((unsigned short*)&lVt[kp * 2][0])[r]     = (unsigned short)bf16_rne(v.x);
      ((unsigned short*)&lVt[kp * 2 + 1][0])[r] = (unsigned short)bf16_rne(v.y);
    }
    __syncthreads();

    // ---- S = Q K^T : 16(q) x 32(kv), inner 128 -> 2 n-tiles x 4 k-steps
    v8f sacc[2] = {};
#pragma unroll
    for (int kk = 0; kk < 4; ++kk) {
      unsigned int ar[8];
      int m = wave * 16 + l15;
#pragma unroll
      for (int p = 0; p < 8; ++p) {
        int kp = kk * 16 + (p < 4 ? 0 : 8) + half * 4 + (p & 3);
        ar[p] = lQ[m][kp];
      }
      v16bf a = frag8(ar);
#pragma unroll
      for (int nt = 0; nt < 2; ++nt) {
        unsigned int br[8];
        int n = nt * 16 + l15;
#pragma unroll
        for (int p = 0; p < 8; ++p) {
          int kp = kk * 16 + (p < 4 ? 0 : 8) + half * 4 + (p & 3);
          br[p] = lK[n][kp];
        }
        sacc[nt] = WMMA_BF16(a, frag8(br), sacc[nt]);
      }
    }

    // ---- scale + causal mask + row max (reduce across 16-lane half)
    float tilemax[8];
#pragma unroll
    for (int r = 0; r < 8; ++r) {
      int qpos = qw + r + 8 * half;
      float mx = -3.0e38f;
#pragma unroll
      for (int nt = 0; nt < 2; ++nt) {
        int kpos = j0 + nt * 16 + l15;
        float v = sacc[nt][r] * scale;
        v = (kpos <= qpos) ? v : -3.0e38f;
        sacc[nt][r] = v;
        mx = fmaxf(mx, v);
      }
#pragma unroll
      for (int off = 1; off < 16; off <<= 1)
        mx = fmaxf(mx, __shfl_xor(mx, off, 32));
      tilemax[r] = mx;
    }

    // ---- online softmax update
    float alpha[8];
#pragma unroll
    for (int r = 0; r < 8; ++r) {
      float mnew = fmaxf(mrow[r], tilemax[r]);
      alpha[r] = __expf(mrow[r] - mnew);
      mrow[r] = mnew;
    }
#pragma unroll
    for (int r = 0; r < 8; ++r) {
      float rs = 0.0f;
      int m = r + 8 * half;
#pragma unroll
      for (int nt = 0; nt < 2; ++nt) {
        float p = __expf(sacc[nt][r] - mrow[r]);
        rs += p;
        ((unsigned short*)&lP[wave][m][0])[nt * 16 + l15] =
            (unsigned short)bf16_rne(p);
      }
#pragma unroll
      for (int off = 1; off < 16; off <<= 1)
        rs += __shfl_xor(rs, off, 32);
      lrow[r] = lrow[r] * alpha[r] + rs;
    }

    // ---- rescale output accumulators
#pragma unroll
    for (int d = 0; d < 8; ++d)
#pragma unroll
      for (int r = 0; r < 8; ++r) o[d][r] *= alpha[r];

    // wave-local LDS write->read ordering for P restaging
    asm volatile("s_wait_dscnt 0" ::: "memory");

    // ---- P fragment (A layout) from LDS
    unsigned int pr[8];
#pragma unroll
    for (int p = 0; p < 8; ++p) {
      int kp = (p < 4 ? 0 : 8) + half * 4 + (p & 3);
      pr[p] = lP[wave][l15][kp];
    }
    v16bf pf = frag8(pr);

    // ---- O += P V : 8 output d-tiles
#pragma unroll
    for (int d = 0; d < 8; ++d) {
      unsigned int vr[8];
      int n = d * 16 + l15;
#pragma unroll
      for (int p = 0; p < 8; ++p) {
        int kp = (p < 4 ? 0 : 8) + half * 4 + (p & 3);
        vr[p] = lVt[n][kp];
      }
      o[d] = WMMA_BF16(pf, frag8(vr), o[d]);
    }
  }

  // ---- normalize + store as bf16
#pragma unroll
  for (int d = 0; d < 8; ++d)
#pragma unroll
    for (int r = 0; r < 8; ++r) {
      int qpos = qw + r + 8 * half;
      int col = h * HEAD_DIM + d * 16 + l15;
      attn_out[(size_t)qpos * QS + col] =
          (unsigned short)bf16_rne(o[d][r] / lrow[r]);
    }
}

// ---------------------------------------------------------------------------
extern "C" void kernel_launch(void* const* d_in, const int* in_sizes, int n_in,
                              void* d_out, int out_size, void* d_ws, size_t ws_size,
                              hipStream_t stream) {
  (void)n_in; (void)out_size; (void)ws_size;
  const int*   positions = (const int*)d_in[0];
  const float* hidden    = (const float*)d_in[1];
  const float* w_qkv     = (const float*)d_in[2];
  const float* w_o       = (const float*)d_in[3];
  float*       out       = (float*)d_out;
  const int    T         = in_sizes[1] / HIDDEN;  // 2048

  // Workspace layout (bytes, 16B-aligned chunks)
  char* ws = (char*)d_ws;
  unsigned short* hidden_bf = (unsigned short*)ws;                    // T*4096
  ws += (size_t)T * HIDDEN * 2;
  unsigned short* wqkv_bf = (unsigned short*)ws;                      // 6144*4096
  ws += (size_t)QKV_ROW * HIDDEN * 2;
  unsigned short* wo_bf = (unsigned short*)ws;                        // 4096*4096
  ws += (size_t)HIDDEN * QS * 2;
  float* qkv = (float*)ws;                                            // T*6144 f32
  ws += (size_t)T * QKV_ROW * 4;
  unsigned short* attn_bf = (unsigned short*)ws;                      // T*4096

  // 0) one-shot bf16 conversions
  {
    int np;
    np = (T * HIDDEN) / 2;
    cvt_bf16_kernel<<<(np + 255) / 256, 256, 0, stream>>>(hidden, (unsigned int*)hidden_bf, np);
    np = (QKV_ROW * HIDDEN) / 2;
    cvt_bf16_kernel<<<(np + 255) / 256, 256, 0, stream>>>(w_qkv, (unsigned int*)wqkv_bf, np);
    np = (HIDDEN * QS) / 2;
    cvt_bf16_kernel<<<(np + 255) / 256, 256, 0, stream>>>(w_o, (unsigned int*)wo_bf, np);
  }

  // 1) qkv = hidden @ w_qkv^T
  gemm_bf16_nt<<<dim3(QKV_ROW / 128, T / 128), 256, 0, stream>>>(
      hidden_bf, wqkv_bf, qkv, T, QKV_ROW, HIDDEN);

  // 2) RoPE on q and k slices
  int rth = T * (NUM_HEADS + NUM_KV) * (HEAD_DIM / 2);
  rope_kernel<<<(rth + 255) / 256, 256, 0, stream>>>(qkv, positions, T);

  // 3) causal flash attention (bf16 out)
  attn_kernel<<<dim3(NUM_HEADS, T / 64), 128, 0, stream>>>(qkv, attn_bf, T);

  // 4) out = attn @ w_o^T
  gemm_bf16_nt<<<dim3(HIDDEN / 128, T / 128), 256, 0, stream>>>(
      attn_bf, wo_bf, out, T, HIDDEN, HIDDEN);
}
